// DistributedExpertPool_87814901334421
// MI455X (gfx1250) — matrix operand order repaired and checked by reading, not compiled, not run on hardware
//
#include <hip/hip_runtime.h>
#include <math.h>

#define NUM_EXPERTS 8
#define HIDDEN 1024
#define FFN 2048
#define TOKENS 2048

#define TILE_M 64
#define TILE_N 128
#define KC 128           // K chunk staged in LDS per iteration
#define PAD 8            // bf16 row padding (16B) -> aligned b128 LDS fragment loads
#define NTHREADS 256     // 8 waves: 2 (M) x 4 (N), each wave owns a 32x32 register tile

typedef __bf16 bf16_t;
typedef __attribute__((ext_vector_type(16))) __bf16 v16bf;
typedef __attribute__((ext_vector_type(8)))  __bf16 v8bf;
typedef __attribute__((ext_vector_type(4)))  __bf16 v4bf;
typedef __attribute__((ext_vector_type(2)))  __bf16 v2bf;
typedef __attribute__((ext_vector_type(8)))  float  v8f;
typedef __attribute__((ext_vector_type(4)))  float  v4f;

static_assert(KC % 32 == 0, "K chunk must be multiple of WMMA K");
static_assert(TILE_N == 128 && TILE_M == 64, "staging index math assumes 64x128");
static_assert((TILE_M * (KC / 4)) % NTHREADS == 0, "A staging uniform");
static_assert(((KC / 2) * (TILE_N / 4)) % NTHREADS == 0, "B staging uniform");

__global__ void zero_counts_kernel(int* __restrict__ counts) {
  if (threadIdx.x < NUM_EXPERTS) counts[threadIdx.x] = 0;
}

__global__ void route_kernel(const int* __restrict__ eidx, int* __restrict__ counts,
                             int* __restrict__ list) {
  int t = blockIdx.x * blockDim.x + threadIdx.x;
  if (t < TOKENS) {
    int e = eidx[t];
    int slot = atomicAdd(&counts[e], 1);
    list[e * TOKENS + slot] = t;
  }
}

// Load 4 consecutive source elements as bf16x4 (converting if fp32).
static __device__ inline v4bf load4_bf16(const float* p) {
  v4f f = *(const v4f*)p;
  v4bf r; r.x = (bf16_t)f.x; r.y = (bf16_t)f.y; r.z = (bf16_t)f.z; r.w = (bf16_t)f.w;
  return r;
}
static __device__ inline v4bf load4_bf16(const bf16_t* p) { return *(const v4bf*)p; }

// Grouped GEMM over one expert's token list:
//   Dst[token, n] = act( sum_k Asrc[token, k] * W[e, k, n] + Bias[e, n] )
template <bool GELU, typename AT, typename OT>
__global__ __launch_bounds__(NTHREADS) void moe_gemm_kernel(
    const AT* __restrict__ Asrc, const float* __restrict__ W,
    const float* __restrict__ Bias, OT* __restrict__ Dst,
    const int* __restrict__ counts, const int* __restrict__ list,
    int Kdim, int Ndim)
{
  const int mtiles = TOKENS / TILE_M;
  const int e   = blockIdx.x / mtiles;
  const int mt  = blockIdx.x % mtiles;
  const int cnt = counts[e];
  if (mt * TILE_M >= cnt) return;   // uniform: EXEC stays all-ones for WMMA

  __shared__ int       s_tid[TILE_M];
  __shared__ const AT* s_aptr[TILE_M];        // gathered row base pointers
  __shared__ bf16_t    sA[TILE_M][KC + PAD];  // [m][k], K-contiguous
  __shared__ bf16_t    sB[TILE_N][KC + PAD];  // [n][k], transposed, K-contiguous

  const int tid   = threadIdx.x;
  const int lane  = tid & 31;
  const int wave  = tid >> 5;
  const int mbase = (wave >> 2) * 32;   // wave's M sub-tile origin (0 or 32)
  const int nbase = (wave & 3) * 32;    // wave's N sub-tile origin (0/32/64/96)
  const int n0    = blockIdx.y * TILE_N;

  if (tid < TILE_M) {
    int r = mt * TILE_M + tid;
    int tok = list[e * TOKENS + ((r < cnt) ? r : (cnt - 1))];  // pad with valid row
    s_tid[tid]  = tok;
    s_aptr[tid] = Asrc + (size_t)tok * Kdim;
  }
  __syncthreads();

  const float* We = W + (size_t)e * (size_t)Kdim * (size_t)Ndim;

  // Fragment indexing per CDNA5 16-bit layouts (wave32):
  //  A 16x32: lane-half -> K +0/+8; VGPRs 0-3 / 4-7 -> K block +0/+16
  //  B 32x16: lane-half -> K +0/+8; elements 0-7 / 8-15 -> K block +0/+16
  const int l15 = lane & 15;
  const int kgA = (lane >> 4) << 3;   // 0 or 8
  const int kgB = (lane >> 4) << 3;   // 0 or 8

  v8f acc[2][2];
  #pragma unroll
  for (int mi = 0; mi < 2; ++mi)
    #pragma unroll
    for (int ni = 0; ni < 2; ++ni)
      acc[mi][ni] = (v8f){0.f,0.f,0.f,0.f,0.f,0.f,0.f,0.f};

  for (int k0 = 0; k0 < Kdim; k0 += KC) {
    // ---- Stage A: gather rows via pointer table, float4 loads, packed bf16x4 stores
    #pragma unroll
    for (int j = 0; j < (TILE_M * (KC / 4)) / NTHREADS; ++j) {   // 8 iters
      int u  = tid + j * NTHREADS;     // float4 unit
      int m  = u >> 5;                 // KC/4 == 32 units per row
      int kv = (u & 31) << 2;
      *(v4bf*)&sA[m][kv] = load4_bf16(s_aptr[m] + (k0 + kv));
    }
    // ---- Stage B: two adjacent K rows per unit, float4 loads, packed bf16x2 stores
    #pragma unroll
    for (int j = 0; j < ((KC / 2) * (TILE_N / 4)) / NTHREADS; ++j) {  // 8 iters
      int u  = tid + j * NTHREADS;
      int k2 = (u >> 5) << 1;          // TILE_N/4 == 32 units per row-pair
      int nv = (u & 31) << 2;
      const float* p0 = &We[(size_t)(k0 + k2) * Ndim + (n0 + nv)];
      v4f r0 = *(const v4f*)p0;
      v4f r1 = *(const v4f*)(p0 + Ndim);
      #pragma unroll
      for (int l = 0; l < 4; ++l) {
        v2bf pk; pk.x = (bf16_t)r0[l]; pk.y = (bf16_t)r1[l];
        *(v2bf*)&sB[nv + l][k2] = pk;
      }
    }
    // Prefetch head of next weight chunk while we compute
    if (k0 + KC < Kdim) {
      int pk = tid >> 5, pn = (tid & 31) << 2;
      __builtin_prefetch(&We[(size_t)(k0 + KC + pk) * Ndim + (n0 + pn)], 0, 1);
    }
    __syncthreads();

    #pragma unroll
    for (int kk = 0; kk < KC; kk += 32) {
      v16bf af[2], bf[2];
      #pragma unroll
      for (int mi = 0; mi < 2; ++mi) {
        const bf16_t* pa = &sA[mbase + mi * 16 + l15][kk + kgA];
        v8bf lo = *(const v8bf*)pa;
        v8bf hi = *(const v8bf*)(pa + 16);
        af[mi] = __builtin_shufflevector(lo, hi, 0,1,2,3,4,5,6,7,8,9,10,11,12,13,14,15);
      }
      #pragma unroll
      for (int ni = 0; ni < 2; ++ni) {
        const bf16_t* pb = &sB[nbase + ni * 16 + l15][kk + kgB];
        v8bf lo = *(const v8bf*)pb;
        v8bf hi = *(const v8bf*)(pb + 16);
        bf[ni] = __builtin_shufflevector(lo, hi, 0,1,2,3,4,5,6,7,8,9,10,11,12,13,14,15);
      }
      #pragma unroll
      for (int mi = 0; mi < 2; ++mi)
        #pragma unroll
        for (int ni = 0; ni < 2; ++ni)
          acc[mi][ni] = __builtin_amdgcn_wmma_f32_16x16x32_bf16(
              false, af[mi], false, bf[ni], (short)0, acc[mi][ni], false, false);
    }
    __syncthreads();
  }

  // Epilogue: bias + (optional exact GELU) in fp32; scatter by token id.
  // C/D layout: VGPR r -> M = r + 8*(lane>=16); N = lane&15.
  const int mrow = mbase + ((lane >> 4) << 3);
  #pragma unroll
  for (int ni = 0; ni < 2; ++ni) {
    const int   ncol = n0 + nbase + ni * 16 + l15;
    const float bv   = Bias[e * Ndim + ncol];
    #pragma unroll
    for (int mi = 0; mi < 2; ++mi) {
      #pragma unroll
      for (int r = 0; r < 8; ++r) {
        int ml = mrow + mi * 16 + r;
        int gm = mt * TILE_M + ml;
        if (gm < cnt) {
          float v = acc[mi][ni][r] + bv;
          if (GELU) v = 0.5f * v * (1.0f + erff(v * 0.70710678118654752f));
          Dst[(size_t)s_tid[ml] * Ndim + ncol] = (OT)v;
        }
      }
    }
  }
}

extern "C" void kernel_launch(void* const* d_in, const int* in_sizes, int n_in,
                              void* d_out, int out_size, void* d_ws, size_t ws_size,
                              hipStream_t stream) {
  (void)in_sizes; (void)n_in; (void)out_size; (void)ws_size;

  const float* x    = (const float*)d_in[0];
  const int*   eidx = (const int*)d_in[1];
  const float* W1   = (const float*)d_in[2];
  const float* b1   = (const float*)d_in[3];
  const float* W2   = (const float*)d_in[4];
  const float* b2   = (const float*)d_in[5];
  float*       out  = (float*)d_out;

  // Workspace: counts[8] | list[8][2048] | h bf16 [TOKENS][FFN] (~8.5 MB total)
  char* ws = (char*)d_ws;
  int*    counts = (int*)ws;
  int*    list   = (int*)(ws + 256);
  bf16_t* h      = (bf16_t*)(ws + 256 + NUM_EXPERTS * TOKENS * sizeof(int));

  zero_counts_kernel<<<1, 64, 0, stream>>>(counts);
  route_kernel<<<TOKENS / 256, 256, 0, stream>>>(eidx, counts, list);

  dim3 g1(NUM_EXPERTS * (TOKENS / TILE_M), FFN / TILE_N);     // (256, 16)
  moe_gemm_kernel<true, float, bf16_t><<<g1, NTHREADS, 0, stream>>>(
      x, W1, b1, h, counts, list, HIDDEN, FFN);

  dim3 g2(NUM_EXPERTS * (TOKENS / TILE_M), HIDDEN / TILE_N);  // (256, 8)
  moe_gemm_kernel<false, bf16_t, float><<<g2, NTHREADS, 0, stream>>>(
      h, W2, b2, out, counts, list, FFN, HIDDEN);
}
